// CombinedRepeatCausalLinear_12481174962957
// MI455X (gfx1250) — matrix operand order repaired and checked by reading, not compiled, not run on hardware
//
#include <hip/hip_runtime.h>

// ---------------------------------------------------------------------------
// out[m, t] = sum_s x[m, s] * W[s, t] + bias[t],  m = flattened (B,E)
// W[s,t] = (s<=t) ? w0[s]*dv0^(t-s) + w1[t]*dv1^(t-s) : 0   (DECAY_CONSTANT=1)
// GEMM M=8192, K=S=2048, N=S=2048, fp32.
//
// Pipeline:
//   k0: split x -> f16 hi/lo (once; removes cvt work from the GEMM hot loop)
//   k1: build W^T -> f16 hi/lo (causal zeros baked in, rows contiguous in K)
//   k2: WMMA GEMM, 3 products (xh*Wh + xh*Wl + xl*Wh) in fp32 accum,
//       double-buffered LDS staged with gfx1250 async global->LDS copies.
// ---------------------------------------------------------------------------

typedef _Float16 vh4  __attribute__((ext_vector_type(4)));
typedef _Float16 vh8  __attribute__((ext_vector_type(8)));
typedef _Float16 vh16 __attribute__((ext_vector_type(16)));
typedef float    vf4  __attribute__((ext_vector_type(4)));
typedef float    vf8  __attribute__((ext_vector_type(8)));
typedef int      vi4  __attribute__((ext_vector_type(4)));

#define BM 128
#define BN 64
#define BK 32
#define LDSS 40              // LDS row stride in halfs: 80B, 16B-aligned, conflict-free
#define XH_OFF 0
#define XL_OFF (BM * LDSS)               // 5120
#define WH_OFF (2 * BM * LDSS)           // 10240
#define WL_OFF (2 * BM * LDSS + BN * LDSS)
#define BUFH   (2 * BM * LDSS + 2 * BN * LDSS)   // 15360 halfs per buffer

#if __has_builtin(__builtin_amdgcn_global_load_async_to_lds_b128)
#define HAVE_ASYNC_LDS 1
#else
#define HAVE_ASYNC_LDS 0
#endif

// pointer types matching the builtin signature: (int4 AS1*, int4 AS3*, Ii, Ii)
typedef __attribute__((address_space(1))) vi4* gptr_vi4;
typedef __attribute__((address_space(3))) vi4* lptr_vi4;

__device__ __forceinline__ void copy16_g2l(_Float16* lds_dst, const _Float16* g_src) {
#if HAVE_ASYNC_LDS
  __builtin_amdgcn_global_load_async_to_lds_b128(
      (gptr_vi4)(g_src), (lptr_vi4)(lds_dst), 0, 0);
#else
  *(vh8*)lds_dst = *(const vh8*)g_src;
#endif
}

__device__ __forceinline__ void wait_async_copies() {
#if HAVE_ASYNC_LDS
#if __has_builtin(__builtin_amdgcn_s_wait_asynccnt)
  __builtin_amdgcn_s_wait_asynccnt(0);
#else
  asm volatile("s_wait_asynccnt 0" ::: "memory");
#endif
#endif
}

// ---------------- Kernel 0: split x into f16 hi/lo -------------------------
__global__ __launch_bounds__(256)
void split_x_kernel(const float* __restrict__ x,
                    _Float16* __restrict__ xh,
                    _Float16* __restrict__ xl,
                    size_t n4) {
  size_t i = (size_t)blockIdx.x * 256 + threadIdx.x;
  if (i >= n4) return;
  vf4 v = *(const vf4*)(x + i * 4);
  vh4 h, l;
  h.x = (_Float16)v.x; l.x = (_Float16)(v.x - (float)h.x);
  h.y = (_Float16)v.y; l.y = (_Float16)(v.y - (float)h.y);
  h.z = (_Float16)v.z; l.z = (_Float16)(v.z - (float)h.z);
  h.w = (_Float16)v.w; l.w = (_Float16)(v.w - (float)h.w);
  *(vh4*)(xh + i * 4) = h;
  *(vh4*)(xl + i * 4) = l;
}

// ---------------- Kernel 1: build W transposed, split into f16 hi/lo -------
__global__ __launch_bounds__(256)
void build_w_kernel(const float* __restrict__ weight,   // [2,S]
                    const float* __restrict__ decay,    // [2,1]
                    _Float16* __restrict__ Wh,          // [S][S]  Wh[t*S+s]
                    _Float16* __restrict__ Wl,
                    int S) {
  int s = blockIdx.x * 256 + threadIdx.x;
  int t = blockIdx.y;
  if (s >= S) return;
  float val = 0.0f;
  if (s <= t) {
    float dv0 = fminf(fmaxf(decay[0], 0.9f), 1.0f);
    float dv1 = fminf(fmaxf(decay[1], 0.9f), 1.0f);
    float e   = (float)(t - s);
    float p0  = __expf(e * __logf(dv0));
    float p1  = __expf(e * __logf(dv1));
    val = weight[s] * p0 + weight[S + t] * p1;
  }
  _Float16 h = (_Float16)val;
  _Float16 l = (_Float16)(val - (float)h);
  size_t g = (size_t)t * S + s;   // transposed: row t contiguous in s (=K)
  Wh[g] = h;
  Wl[g] = l;
}

// ---------------- tile staging (global f16 -> LDS, 16B per lane) -----------
__device__ __forceinline__ void stage_tile(int tid, int m0, int n0, int k0, int S,
                                           const _Float16* __restrict__ xh,
                                           const _Float16* __restrict__ xl,
                                           const _Float16* __restrict__ Wh,
                                           const _Float16* __restrict__ Wl,
                                           _Float16* buf) {
  // x tiles: BM x BK halfs = 512 16B chunks per array, 2 per thread
#pragma unroll
  for (int i = 0; i < 2; ++i) {
    int idx = tid + i * 256;
    int r   = idx >> 2;          // 4 chunks per 32-half row
    int c   = (idx & 3) * 8;
    size_t g = (size_t)(m0 + r) * S + (k0 + c);
    copy16_g2l(&buf[XH_OFF + r * LDSS + c], xh + g);
    copy16_g2l(&buf[XL_OFF + r * LDSS + c], xl + g);
  }
  // W tiles: BN x BK halfs = 256 16B chunks per array, 1 per thread
  {
    int n = tid >> 2;
    int c = (tid & 3) * 8;
    size_t g = (size_t)(n0 + n) * S + (k0 + c);
    copy16_g2l(&buf[WH_OFF + n * LDSS + c], Wh + g);
    copy16_g2l(&buf[WL_OFF + n * LDSS + c], Wl + g);
  }
}

// ---------------- Kernel 2: WMMA GEMM --------------------------------------
__global__ __launch_bounds__(256)
void gemm_wmma_kernel(const _Float16* __restrict__ xh,  // [M][S] f16 hi
                      const _Float16* __restrict__ xl,  // [M][S] f16 lo
                      const _Float16* __restrict__ Wh,  // [S][S] transposed
                      const _Float16* __restrict__ Wl,
                      const float* __restrict__ bias,   // [S]
                      float* __restrict__ out,          // [M][S]
                      int M, int S) {
  __shared__ _Float16 smem[2 * BUFH];    // double buffer, 2 x 30 KB

  const int tid  = threadIdx.x;
  const int lane = tid & 31;
  const int wave = tid >> 5;
  const int n0 = blockIdx.x * BN;
  const int m0 = blockIdx.y * BM;

  const int waveM = (wave & 3) * 32;   // 4 waves along M
  const int waveN = (wave >> 2) * 32;  // 2 waves along N

  const int l15    = lane & 15;
  const int hiHalf = lane >> 4;            // 0 or 1
  const int kA0 = hiHalf ? 8 : 0;          // A: halves 0..7 -> K=kA0.., 8..15 -> K=kA0+16..
  const int kB0 = hiHalf ? 8 : 0;          // B: same K order as A

  vf8 acc[2][2] = {};

  // Causal: W[s,t]==0 for s>t, max t in this block = n0+BN-1.
  const int ksteps = min(S, n0 + BN) / BK;

  // prefetch tile 0 into buffer 0
  stage_tile(tid, m0, n0, 0, S, xh, xl, Wh, Wl, smem);

  for (int kt = 0; kt < ksteps; ++kt) {
    _Float16* cur = smem + (size_t)(kt & 1) * BUFH;
    _Float16* nxt = smem + (size_t)((kt + 1) & 1) * BUFH;

    wait_async_copies();     // this wave's copies of tile kt are done
    __syncthreads();         // everyone's tile kt is visible

    if (kt + 1 < ksteps) {
      stage_tile(tid, m0, n0, (kt + 1) * BK, S, xh, xl, Wh, Wl, nxt);
    }

    // ---- load fragments per ISA 7.12.2 layouts ----
    vh16 Ah[2], Al[2], Bh[2], Bl[2];
#pragma unroll
    for (int tm = 0; tm < 2; ++tm) {
      const _Float16* bh = &cur[XH_OFF + (waveM + tm * 16 + l15) * LDSS];
      const _Float16* bl = &cur[XL_OFF + (waveM + tm * 16 + l15) * LDSS];
      vh8 a0 = *(const vh8*)(bh + kA0);
      vh8 a1 = *(const vh8*)(bh + kA0 + 16);
      Ah[tm] = __builtin_shufflevector(a0, a1, 0,1,2,3,4,5,6,7,8,9,10,11,12,13,14,15);
      vh8 b0 = *(const vh8*)(bl + kA0);
      vh8 b1 = *(const vh8*)(bl + kA0 + 16);
      Al[tm] = __builtin_shufflevector(b0, b1, 0,1,2,3,4,5,6,7,8,9,10,11,12,13,14,15);
    }
#pragma unroll
    for (int tn = 0; tn < 2; ++tn) {
      const _Float16* bh = &cur[WH_OFF + (waveN + tn * 16 + l15) * LDSS + kB0];
      const _Float16* bl = &cur[WL_OFF + (waveN + tn * 16 + l15) * LDSS + kB0];
      vh8 c0 = *(const vh8*)(bh);
      vh8 c1 = *(const vh8*)(bh + 16);
      Bh[tn] = __builtin_shufflevector(c0, c1, 0,1,2,3,4,5,6,7,8,9,10,11,12,13,14,15);
      vh8 d0 = *(const vh8*)(bl);
      vh8 d1 = *(const vh8*)(bl + 16);
      Bl[tn] = __builtin_shufflevector(d0, d1, 0,1,2,3,4,5,6,7,8,9,10,11,12,13,14,15);
    }

    // ---- 3-product f16 split GEMM: xh*Wh + xh*Wl + xl*Wh, fp32 accum ----
#pragma unroll
    for (int tm = 0; tm < 2; ++tm) {
#pragma unroll
      for (int tn = 0; tn < 2; ++tn) {
        acc[tm][tn] = __builtin_amdgcn_wmma_f32_16x16x32_f16(
            false, Ah[tm], false, Bh[tn], (short)0, acc[tm][tn], false, false);
        acc[tm][tn] = __builtin_amdgcn_wmma_f32_16x16x32_f16(
            false, Ah[tm], false, Bl[tn], (short)0, acc[tm][tn], false, false);
        acc[tm][tn] = __builtin_amdgcn_wmma_f32_16x16x32_f16(
            false, Al[tm], false, Bh[tn], (short)0, acc[tm][tn], false, false);
      }
    }
    __syncthreads();   // all waves done reading `cur` before it is re-staged
  }

  // ---- epilogue: C layout VGPR r -> (M = r + 8*hiHalf, N = lane&15) ----
#pragma unroll
  for (int tm = 0; tm < 2; ++tm) {
#pragma unroll
    for (int tn = 0; tn < 2; ++tn) {
      int gm = m0 + waveM + tm * 16 + hiHalf * 8;
      int gn = n0 + waveN + tn * 16 + l15;
      float bv = bias[gn];
#pragma unroll
      for (int r = 0; r < 8; ++r) {
        out[(size_t)(gm + r) * S + gn] = acc[tm][tn][r] + bv;
      }
    }
  }
}

// ---------------------------------------------------------------------------
extern "C" void kernel_launch(void* const* d_in, const int* in_sizes, int n_in,
                              void* d_out, int out_size, void* d_ws, size_t ws_size,
                              hipStream_t stream) {
  const float* x      = (const float*)d_in[0];   // [B,E,S]
  const float* weight = (const float*)d_in[1];   // [2,S]
  const float* bias   = (const float*)d_in[2];   // [S]
  const float* decay  = (const float*)d_in[3];   // [2,1]

  const int S = in_sizes[2];                 // 2048
  const int M = in_sizes[0] / S;             // B*E = 8192

  // workspace layout: xh, xl ([M][S] f16 each), Wh, Wl ([S][S] f16 each)
  _Float16* xh = (_Float16*)d_ws;
  _Float16* xl = xh + (size_t)M * S;
  _Float16* Wh = xl + (size_t)M * S;
  _Float16* Wl = Wh + (size_t)S * S;

  size_t n4 = (size_t)M * S / 4;
  split_x_kernel<<<dim3((unsigned)((n4 + 255) / 256)), 256, 0, stream>>>(x, xh, xl, n4);

  dim3 g1((S + 255) / 256, S);
  build_w_kernel<<<g1, 256, 0, stream>>>(weight, decay, Wh, Wl, S);

  dim3 g2(S / BN, M / BM);                   // (32, 64)
  gemm_wmma_kernel<<<g2, 256, 0, stream>>>(xh, xl, Wh, Wl, bias, (float*)d_out, M, S);
}
